// QRNN_5703716569536
// MI455X (gfx1250) — matrix operand order repaired and checked by reading, compiled, not run on hardware
//
#include <hip/hip_runtime.h>
#include <hip/hip_bf16.h>
#include <math.h>

// ---------- CDNA5 WMMA types / helpers ----------
typedef _Float16 v16h __attribute__((ext_vector_type(16)));
typedef _Float16 h8   __attribute__((ext_vector_type(8)));
typedef float    v8f  __attribute__((ext_vector_type(8)));

#define WMMA_F16(a, b, c) \
  __builtin_amdgcn_wmma_f32_16x16x32_f16(false, (a), false, (b), (short)0, (c), false, false)

// A-fragment (16x32 f16, row-major source): lane m=lane&15, half-wave K-base 0/8,
// elements 0..7 -> K=kb..kb+7, elements 8..15 -> K=kb+16..kb+23  (ISA 7.12.2)
__device__ __forceinline__ v16h ldA16(const _Float16* p) {
  union { v16h v; h8 h[2]; } u;
  u.h[0] = *reinterpret_cast<const h8*>(p);
  u.h[1] = *reinterpret_cast<const h8*>(p + 16);
  return u.v;
}
// B-fragment (32x16 f16): lane n=lane&15, half-wave holds K 0..15 / 16..31 -> one 32B read.
__device__ __forceinline__ v16h ldB16(const void* p) {
  return *reinterpret_cast<const v16h*>(p);
}

// async global -> LDS copy, 16B per lane (gfx1250 GLOBAL_LOAD_ASYNC_TO_LDS_B128, ASYNCcnt)
__device__ __forceinline__ void async_cp16(void* lds, const void* gptr) {
  asm volatile("global_load_async_to_lds_b128 %0, %1, off"
               :
               : "v"((unsigned)(size_t)lds), "v"(gptr)
               : "memory");
}
#define WAIT_ASYNC(n) asm volatile("s_wait_asynccnt " #n ::: "memory")

// ---------- fp32 -> fp16 convert (grid-stride) ----------
__global__ __launch_bounds__(256) void f32_to_f16(const float* __restrict__ src,
                                                  _Float16* __restrict__ dst, size_t n) {
  size_t i = (size_t)blockIdx.x * blockDim.x + threadIdx.x;
  size_t stride = (size_t)gridDim.x * blockDim.x;
  for (; i < n; i += stride) dst[i] = (_Float16)src[i];
}

// ---------- QRNN layer gate GEMM ----------
// Y[s,b,:] = concat(cur[s,b,:], cur[s-1,b,:]) @ W.T ; W is (3072, 2*IN) f16 row-major.
// Block = 4 waves = 64 rows (one timestep s) x 64 cols. Weights staged through LDS with
// triple-buffered async copies (chunk = 64 K); activations read direct from global (L2).
// Epilogue fuses tanh/sigmoid and stores f16 gates.
#define CHUNK_K   64
#define ROW_BYTES 128                 // CHUNK_K * 2
#define ROW_PITCH 144                 // +16B pad: bank-conflict-free fragment reads
#define BUF_BYTES (64 * ROW_PITCH)    // 9216
__global__ __launch_bounds__(128) void qrnn_gemm(
    const _Float16* __restrict__ cur,  // (S,B,IN) f16
    const _Float16* __restrict__ W,    // (3072, 2*IN) f16
    _Float16* __restrict__ Zb, _Float16* __restrict__ Fb, _Float16* __restrict__ Gb,
    int IN)
{
  __shared__ char smem[3 * BUF_BYTES];   // 27648 B of 320KB/WGP

  const int lane = threadIdx.x;
  const int wave = threadIdx.y;                  // 0..3
  const int M0 = (blockIdx.y * 4 + wave) * 16;   // row = s*64 + b
  const int N0 = blockIdx.x * 64;
  const int s  = blockIdx.y;                     // block covers exactly one timestep
  const int b0 = M0 & 63;
  const int K2 = 2 * IN;

  const int m   = lane & 15;
  const int kbA = (lane >> 4) * 16;
  const int kbB = (lane >> 4) * 32;              // byte offset of half-wave K group in LDS row

  const _Float16* a0 = cur + ((size_t)s * 64 + b0 + m) * IN + kbA;                    // x_t
  const _Float16* a1 = cur + ((size_t)(s > 0 ? s - 1 : 0) * 64 + b0 + m) * IN + kbA;  // x_{t-1}

  // ---- async staging setup: 64 rows x 128B per chunk, 2 threads/row, 4x16B per thread
  const int tid   = wave * 32 + lane;
  const int srow  = tid >> 1;
  const int shalf = (tid & 1) * 64;
  char* lrow = smem + srow * ROW_PITCH + shalf;
  const char* grow = (const char*)(W + (size_t)(N0 + srow) * K2) + shalf;

  const int nchunk = ((s > 0) ? K2 : IN) / CHUNK_K;   // second K half is zero when s==0

  // prologue: stage chunks 0 and 1
  {
    char* l0 = lrow;                       // buffer 0
    const char* g0 = grow;
#pragma unroll
    for (int i = 0; i < 4; ++i) async_cp16(l0 + i * 16, g0 + i * 16);
    char* l1 = lrow + BUF_BYTES;           // buffer 1
    const char* g1 = grow + ROW_BYTES;
#pragma unroll
    for (int i = 0; i < 4; ++i) async_cp16(l1 + i * 16, g1 + i * 16);
  }

  v8f acc[4] = {};

  for (int kt = 0; kt < nchunk; ++kt) {
    // wait for own portion of chunk kt (lookahead keeps at most 2 chunks = 8 copies in flight)
    if (kt < nchunk - 1) { WAIT_ASYNC(4); } else { WAIT_ASYNC(0); }
    __syncthreads();                       // all waves' portions of chunk kt now in LDS
    if (kt + 2 < nchunk) {                 // stage chunk kt+2 (buffer read last at step kt-1)
      char* l = smem + ((kt + 2) % 3) * BUF_BYTES + srow * ROW_PITCH + shalf;
      const char* g = grow + (size_t)(kt + 2) * ROW_BYTES;
#pragma unroll
      for (int i = 0; i < 4; ++i) async_cp16(l + i * 16, g + i * 16);
    }

    const char* lbuf = smem + (kt % 3) * BUF_BYTES;
#pragma unroll
    for (int j = 0; j < 2; ++j) {          // 2 K-steps of 32 per chunk
      const int k = kt * CHUNK_K + j * 32; // global K index
      v16h a = (k < IN) ? ldB16(a0 + k) : ldB16(a1 + (k - IN));
#pragma unroll
      for (int nt = 0; nt < 4; ++nt) {
        v16h b = ldB16(lbuf + (nt * 16 + m) * ROW_PITCH + j * 64 + kbB);
        acc[nt] = WMMA_F16(a, b, acc[nt]);
      }
    }
  }

  // epilogue: gate activation + f16 store. 64-wide tile lies in a single gate (64 | 1024).
  const int gate  = N0 >> 10;            // 0=Z,1=F,2=Og
  const int nbase = N0 & 1023;
  _Float16* outb = (gate == 0) ? Zb : ((gate == 1) ? Fb : Gb);
  const int mm_hi = (lane >> 4) * 8;     // C/D: VGPR v -> row v (lanes0-15) / v+8 (lanes16-31)
#pragma unroll
  for (int nt = 0; nt < 4; ++nt) {
    const int n = nbase + nt * 16 + m;
#pragma unroll
    for (int v = 0; v < 8; ++v) {
      const int row = b0 + v + mm_hi;
      float val = acc[nt][v];
      float r = (gate == 0) ? tanhf(val) : 1.0f / (1.0f + expf(-val));
      outb[((size_t)s * 64 + row) * 1024 + n] = (_Float16)r;
    }
  }
}

// ---------- fo-pool scan (sequential over S, parallel over B*H) ----------
__global__ __launch_bounds__(256) void qrnn_scan(
    const _Float16* __restrict__ Z, const _Float16* __restrict__ F,
    const _Float16* __restrict__ G, const float* __restrict__ c0,
    _Float16* __restrict__ out)
{
  const int idx = blockIdx.x * blockDim.x + threadIdx.x;   // 0 .. B*H-1
  float c = c0[idx];
  size_t off = (size_t)idx;
  const size_t stride = (size_t)64 * 1024;
  for (int s = 0; s < 512; ++s, off += stride) {
    float f = (float)F[off];
    float z = (float)Z[off];
    c = f * c + (1.0f - f) * z;
    float g = (float)G[off];
    out[off] = (_Float16)(g * c);
  }
}

// ---------- small FC GEMM (M=64 fixed): out = [relu](A @ W.T + bias) ----------
__global__ __launch_bounds__(128) void fc_gemm(
    const _Float16* __restrict__ A, int lda,
    const _Float16* __restrict__ W, int K, int N,
    const float* __restrict__ bias,
    _Float16* __restrict__ outh, float* __restrict__ outf, int relu)
{
  const int lane = threadIdx.x;
  const int M0 = threadIdx.y * 16;      // 4 waves cover M=64
  const int N0 = blockIdx.x * 64;

  const int m   = lane & 15;
  const int kbA = (lane >> 4) * 16;
  const int kbB = (lane >> 4) * 16;

  const _Float16* ap = A + (size_t)(M0 + m) * lda + kbA;
  const _Float16* bp = W + (size_t)(N0 + m) * K + kbB;

  v8f acc[4] = {};
  for (int k = 0; k < K; k += 32) {
    v16h a = ldB16(ap + k);
#pragma unroll
    for (int nt = 0; nt < 4; ++nt) {
      v16h b = ldB16(bp + (size_t)nt * 16 * K + k);
      acc[nt] = WMMA_F16(a, b, acc[nt]);
    }
  }

  const int mm_hi = (lane >> 4) * 8;
#pragma unroll
  for (int nt = 0; nt < 4; ++nt) {
    const int n = N0 + nt * 16 + m;
#pragma unroll
    for (int v = 0; v < 8; ++v) {
      const int row = M0 + v + mm_hi;
      float val = acc[nt][v] + bias[n];
      if (relu) val = fmaxf(val, 0.0f);
      if (outh) outh[(size_t)row * N + n] = (_Float16)val;
      else      outf[(size_t)row * N + n] = val;
    }
  }
}

// ---------------------------------------------------------------------------
extern "C" void kernel_launch(void* const* d_in, const int* in_sizes, int n_in,
                              void* d_out, int out_size, void* d_ws, size_t ws_size,
                              hipStream_t stream) {
  (void)in_sizes; (void)n_in; (void)out_size; (void)ws_size;
  const float* x    = (const float*)d_in[0];   // (512,64,512)
  const float* h    = (const float*)d_in[1];   // (3,64,1024)
  const float* Wl[3] = { (const float*)d_in[2], (const float*)d_in[3], (const float*)d_in[4] };
  const float* Wfc1 = (const float*)d_in[5];   // (1024,1024)
  const float* bfc1 = (const float*)d_in[6];   // (1024,)
  const float* Wfc2 = (const float*)d_in[7];   // (256,1024)
  const float* bfc2 = (const float*)d_in[8];   // (256,)
  float* out = (float*)d_out;                  // (64,256)

  char* ws = (char*)d_ws;
  const size_t SBH2 = (size_t)512 * 64 * 1024 * 2;          // bytes of (S,B,H) f16
  _Float16* bufA = (_Float16*)(ws);
  _Float16* bufB = (_Float16*)(ws + SBH2);
  _Float16* Zb   = (_Float16*)(ws + 2 * SBH2);
  _Float16* Fb   = (_Float16*)(ws + 3 * SBH2);
  _Float16* Gb   = (_Float16*)(ws + 4 * SBH2);
  _Float16* Wh   = (_Float16*)(ws + 5 * SBH2);               // up to 3072*4096 f16 = 25.2MB
  _Float16* hidh = (_Float16*)(ws + 5 * SBH2 + (size_t)3072 * 4096 * 2);

  f32_to_f16<<<2048, 256, 0, stream>>>(x, bufA, (size_t)512 * 64 * 512);

  _Float16* cur = bufA;
  _Float16* nxt = bufB;
  const int INs[3] = { 512, 1024, 1024 };
  for (int l = 0; l < 3; ++l) {
    const int IN = INs[l];
    f32_to_f16<<<2048, 256, 0, stream>>>(Wl[l], Wh, (size_t)3072 * 2 * IN);
    qrnn_gemm<<<dim3(48, 512), dim3(32, 4), 0, stream>>>(cur, Wh, Zb, Fb, Gb, IN);
    qrnn_scan<<<(64 * 1024) / 256, 256, 0, stream>>>(Zb, Fb, Gb,
                                                     h + (size_t)l * 64 * 1024, nxt);
    _Float16* t = cur; cur = nxt; nxt = t;
  }

  const _Float16* last = cur + (size_t)511 * 64 * 1024;      // (64,1024) f16
  f32_to_f16<<<1024, 256, 0, stream>>>(Wfc1, Wh, (size_t)1024 * 1024);
  fc_gemm<<<dim3(16, 1), dim3(32, 4), 0, stream>>>(last, 1024, Wh, 1024, 1024,
                                                   bfc1, hidh, nullptr, 1);
  f32_to_f16<<<256, 256, 0, stream>>>(Wfc2, Wh, (size_t)256 * 1024);
  fc_gemm<<<dim3(4, 1), dim3(32, 4), 0, stream>>>(hidh, 1024, Wh, 1024, 256,
                                                  bfc2, nullptr, out, 0);
}
